// Directional_SAttention_86543591015033
// MI455X (gfx1250) — matrix operand order repaired and checked by reading, not compiled, not run on hardware
//
#include <hip/hip_runtime.h>
#include <math.h>

// ---------------------------------------------------------------------------
// Problem constants
// ---------------------------------------------------------------------------
#define NN   512        // N
#define NM1  511        // N-1
#define EE   64         // E
#define HID  256
#define OUTD 32
#define MPAIR (NN * NM1)      // 261632 off-diagonal pairs
#define WAVES_PER_BLOCK 8
#define ROWS_PER_WAVE   16
#define MAIN_BLOCKS (MPAIR / (WAVES_PER_BLOCK * ROWS_PER_WAVE))  // 2044

typedef __attribute__((ext_vector_type(16))) _Float16 v16h;
typedef __attribute__((ext_vector_type(8)))  _Float16 v8h;
typedef __attribute__((ext_vector_type(8)))  float    v8f;

// Workspace layout (bytes)
#define OFF_BGT   0u                                   // [256][128] f16 = 65536
#define OFF_BKVT  (OFF_BGT  + 256u*128u*2u)            // [128][64]  f16 = 16384
#define OFF_HS    (OFF_BKVT + 128u*64u*2u)             // [512][64]  f32 = 131072
#define OFF_Q0    (OFF_HS   + 512u*64u*4u)             // [512][64]  f32 = 131072
#define OFF_K2    (OFF_Q0   + 512u*64u*4u)             // [512][512][64] f32 = 64 MiB
#define OFF_V2    (OFF_K2   + 512ull*512ull*64ull*4ull)

// ---------------------------------------------------------------------------
// WMMA helpers (CDNA5 v_wmma_f32_16x16x32_f16)
// ---------------------------------------------------------------------------
__device__ __forceinline__ v8f wmma16(v16h a, v16h b, v8f c) {
  return __builtin_amdgcn_wmma_f32_16x16x32_f16(false, a, false, b,
                                                (short)0, c, false, false);
}

// A fragment (16x32 f16) from LDS row-major [16][stride].
// Lane l<16 holds row l, K = {kc..kc+7, kc+16..kc+23};
// lane l>=16 holds row l-16, K = {kc+8..kc+15, kc+24..kc+31}.
__device__ __forceinline__ v16h load_a_lds(const _Float16* base, int stride,
                                           int kc, int lane) {
  int h = lane >> 4, m = lane & 15;
  const _Float16* p = base + m * stride + kc + 8 * h;
  union { v16h v; v8h q[2]; } u;
  u.q[0] = *(const v8h*)(p);
  u.q[1] = *(const v8h*)(p + 16);
  return u.v;
}

// B fragment (32x16 f16) from global N-major f16 matrix BT[ntot][ktot].
// Lane l holds column n0+(l&15); K = {kc+8h..kc+8h+7, kc+16+8h..kc+16+8h+7}, h = l>>4.
__device__ __forceinline__ v16h load_b_glb(const _Float16* BT, int ktot,
                                           int n0, int kc, int lane) {
  int h = lane >> 4, col = lane & 15;
  const _Float16* p = BT + (size_t)(n0 + col) * ktot + kc + 8 * h;
  union { v16h v; v8h q[2]; } u;
  u.q[0] = *(const v8h*)(p);
  u.q[1] = *(const v8h*)(p + 16);
  return u.v;
}

__device__ __forceinline__ float sigm(float x) {
  return 1.0f / (1.0f + __expf(-x));
}

// ---------------------------------------------------------------------------
// Kernel 1: prep — hs, folded weights, f16 transposed B matrices, s=0 rows
// ---------------------------------------------------------------------------
__global__ void dsa_prep(const float* __restrict__ hidden_state,
                         const float* __restrict__ W_hs, const float* __restrict__ b_hs,
                         const float* __restrict__ W_ih, const float* __restrict__ W_hh,
                         const float* __restrict__ Wq,  const float* __restrict__ Wk,
                         const float* __restrict__ Wv,  const float* __restrict__ W_in,
                         const float* __restrict__ b_in,
                         _Float16* __restrict__ BgT, _Float16* __restrict__ BkvT,
                         float* __restrict__ hs, float* __restrict__ q0,
                         float* __restrict__ k2ws, float* __restrict__ v2ws) {
  __shared__ float Wqc[64 * 64];
  const int t = threadIdx.x;

  // hs = hidden_state @ W_hs^T + b_hs   [512][64]
  for (int idx = t; idx < NN * EE; idx += 256) {
    int b = idx >> 6, e = idx & 63;
    float a = b_hs[e];
    const float* hp = hidden_state + (size_t)b * HID;
    const float* wp = W_hs + (size_t)e * HID;
    for (int k = 0; k < HID; ++k) a += wp[k] * hp[k];
    hs[idx] = a;
  }
  // Wqc = Wq2 @ Wq  (W_in rows 0..63)
  for (int idx = t; idx < 64 * 64; idx += 256) {
    int e = idx >> 6, k = idx & 63;
    float a = 0.f;
    for (int x = 0; x < 64; ++x) a += W_in[e * 64 + x] * Wq[x * 64 + k];
    Wqc[idx] = a;
  }
  // BgT[n][k] = (k<64) ? W_ih[n][k] : W_hh[n][k-64]   (f16, N-major 256x128)
  for (int idx = t; idx < 256 * 128; idx += 256) {
    int n = idx >> 7, k = idx & 127;
    float v = (k < 64) ? W_ih[n * 64 + k] : W_hh[n * 64 + (k - 64)];
    BgT[idx] = (_Float16)v;
  }
  // BkvT: rows 0..63 = Wk2@Wk, rows 64..127 = Wv2@Wv   (f16, N-major 128x64)
  for (int idx = t; idx < 128 * 64; idx += 256) {
    int n = idx >> 6, k = idx & 63;
    float a = 0.f;
    if (n < 64)
      for (int x = 0; x < 64; ++x) a += W_in[(64 + n) * 64 + x] * Wk[x * 64 + k];
    else
      for (int x = 0; x < 64; ++x) a += W_in[(128 + (n - 64)) * 64 + x] * Wv[x * 64 + k];
    BkvT[idx] = (_Float16)a;
  }
  __syncthreads();

  // q0 = scale*(hs@Wqc^T + bq2);  k2/v2 rows at s=0 come from hs
  for (int idx = t; idx < NN * EE; idx += 256) {
    int b = idx >> 6, e = idx & 63;
    const float* hp = hs + (size_t)b * EE;
    float aq = b_in[e], ak = b_in[64 + e], av = b_in[128 + e];
    for (int k = 0; k < 64; ++k) {
      float h = hp[k];
      aq += Wqc[e * 64 + k] * h;
      ak += (float)BkvT[e * 64 + k] * h;
      av += (float)BkvT[(64 + e) * 64 + k] * h;
    }
    q0[idx] = aq * 0.125f;                          // 1/sqrt(64)
    k2ws[(size_t)b * EE + e] = ak;                  // s = 0
    v2ws[(size_t)b * EE + e] = av;                  // s = 0
  }
}

// ---------------------------------------------------------------------------
// Kernel 2: main — fused embed + LSTM-gate WMMA + elementwise + k2/v2 WMMA
// ---------------------------------------------------------------------------
__global__ void __launch_bounds__(256)
dsa_main(const float* __restrict__ obs1, const float* __restrict__ obs2,
         const float* __restrict__ h0,   const float* __restrict__ c0,
         const float* __restrict__ W_s,  const float* __restrict__ b_s,
         const float* __restrict__ W_v,  const float* __restrict__ b_v,
         const float* __restrict__ b_ih, const float* __restrict__ b_hh,
         const float* __restrict__ b_in,
         const _Float16* __restrict__ BgT, const _Float16* __restrict__ BkvT,
         float* __restrict__ k2ws, float* __restrict__ v2ws) {
  __shared__ alignas(16) _Float16 AxS[WAVES_PER_BLOCK][ROWS_PER_WAVE][128]; // 32 KiB
  __shared__ alignas(16) _Float16 AhS[WAVES_PER_BLOCK][ROWS_PER_WAVE][64];  // 16 KiB

  const int w    = threadIdx.x >> 5;
  const int lane = threadIdx.x & 31;
  const int half = lane >> 4;
  const int col  = lane & 15;
  const int tileBase = (blockIdx.x * WAVES_PER_BLOCK + w) * ROWS_PER_WAVE;

  // --- Stage A matrix: x = [spat_embed | vel_embed] (cols 0..63), h (64..127)
  #pragma unroll 4
  for (int r = 0; r < ROWS_PER_WAVE; ++r) {
    int p  = tileBase + r;
    int i  = p / NM1;
    int jj = p - i * NM1;
    int j  = jj + (jj >= i);

    float o2jx = obs2[2 * j], o2jy = obs2[2 * j + 1];
    float o2ix = obs2[2 * i], o2iy = obs2[2 * i + 1];
    float rpx = o2jx - o2ix, rpy = o2jy - o2iy;
    float rvx = 4.0f * ((o2jx - obs1[2 * j])     - (o2ix - obs1[2 * i]));
    float rvy = 4.0f * ((o2jy - obs1[2 * j + 1]) - (o2iy - obs1[2 * i + 1]));

    int e = lane;  // 0..31 : one spat + one vel element per lane
    float sp = fmaxf(0.f, W_s[2 * e] * rpx + W_s[2 * e + 1] * rpy + b_s[e]);
    float ve = fmaxf(0.f, W_v[2 * e] * rvx + W_v[2 * e + 1] * rvy + b_v[e]);
    AxS[w][r][e]      = (_Float16)sp;
    AxS[w][r][32 + e] = (_Float16)ve;

    const float* hrow = h0 + ((size_t)i * NN + j) * EE;
    float2 hv = *(const float2*)(hrow + 2 * lane);
    AxS[w][r][64 + 2 * lane]     = (_Float16)hv.x;
    AxS[w][r][64 + 2 * lane + 1] = (_Float16)hv.y;

    __builtin_prefetch(c0 + ((size_t)i * NN + j) * EE, 0, 1);
  }

  // --- Per-lane row metadata, computed ONCE (lane touches rows m = 8*half+r).
  // All flattened element offsets fit in 32 bits (512*512*64 < 2^31).
  int cbase[8];   // (i*NN + j)*EE : base into c0 (and conceptually h0)
  int obase[8];   // (s*NN + i)*EE : base into k2ws / v2ws
  #pragma unroll
  for (int r = 0; r < 8; ++r) {
    int m  = r + 8 * half;
    int p  = tileBase + m;
    int i  = p / NM1;
    int jj = p - i * NM1;
    int j  = jj + (jj >= i);
    int s  = 1 + jj;
    cbase[r] = (i * NN + j) * EE;
    obase[r] = (s * NN + i) * EE;
  }
  __syncthreads();

  // --- GEMM1: gates[16x256] = A[16x128] @ BgT^T, via 16 N-tiles x 4 K-chunks
  v16h axf[4];
  #pragma unroll
  for (int kc = 0; kc < 4; ++kc)
    axf[kc] = load_a_lds(&AxS[w][0][0], 128, kc * 32, lane);

  #pragma unroll
  for (int tp = 0; tp < 4; ++tp) {            // e-subtile: e = 16*tp + col
    v8f ag[4];                                 // i, f, g, o gate tiles
    #pragma unroll
    for (int g = 0; g < 4; ++g) {
      int tIdx = 4 * g + tp;
      v8f acc = {};
      #pragma unroll
      for (int kc = 0; kc < 4; ++kc) {
        v16h bf = load_b_glb(BgT, 128, tIdx * 16, kc * 32, lane);
        acc = wmma16(axf[kc], bf, acc);
      }
      ag[g] = acc;
    }
    int e = 16 * tp + col;
    float bi = b_ih[e]        + b_hh[e];
    float bf = b_ih[64 + e]   + b_hh[64 + e];
    float bg = b_ih[128 + e]  + b_hh[128 + e];
    float bo = b_ih[192 + e]  + b_hh[192 + e];

    #pragma unroll
    for (int r = 0; r < 8; ++r) {             // D layout: M = r + 8*half, N = col
      int m  = r + 8 * half;
      float c  = c0[(size_t)(cbase[r] + e)];
      float ig = sigm(ag[0][r] + bi);
      float fg = sigm(ag[1][r] + bf);
      float gg = tanhf(ag[2][r] + bg);
      float og = sigm(ag[3][r] + bo);
      float cn = fg * c + ig * gg;
      AhS[w][m][e] = (_Float16)(og * tanhf(cn));
    }
  }
  __syncthreads();

  // --- GEMM2: [k2|v2][16x128] = h_new[16x64] @ BkvT^T
  v16h ahf[2];
  ahf[0] = load_a_lds(&AhS[w][0][0], 64, 0, lane);
  ahf[1] = load_a_lds(&AhS[w][0][0], 64, 32, lane);

  #pragma unroll
  for (int t2 = 0; t2 < 8; ++t2) {
    v8f acc = {};
    #pragma unroll
    for (int kc = 0; kc < 2; ++kc) {
      v16h bf = load_b_glb(BkvT, 64, t2 * 16, kc * 32, lane);
      acc = wmma16(ahf[kc], bf, acc);
    }
    int n = t2 * 16 + col;
    bool isK = (n < 64);
    int  e   = isK ? n : (n - 64);
    float bias = isK ? b_in[64 + e] : b_in[128 + e];
    float* dst = isK ? k2ws : v2ws;

    #pragma unroll
    for (int r = 0; r < 8; ++r) {
      dst[(size_t)(obase[r] + e)] = acc[r] + bias;
    }
  }
}

// ---------------------------------------------------------------------------
// Kernel 3: attention row L=0 + output projections, one block per batch b
// ---------------------------------------------------------------------------
__global__ void __launch_bounds__(256)
dsa_attn(const float* __restrict__ q0, const float* __restrict__ k2ws,
         const float* __restrict__ v2ws,
         const float* __restrict__ W_out, const float* __restrict__ b_out,
         const float* __restrict__ W_o,   const float* __restrict__ b_o,
         float* __restrict__ out) {
  __shared__ float sc[NN];
  __shared__ float red[256];
  __shared__ float part[4][64];
  __shared__ float ctxv[64];
  const int b = blockIdx.x, t = threadIdx.x;

  // scores[s] = q0[b] . k2[s][b]
  const float* qp = q0 + (size_t)b * EE;
  for (int s = t; s < NN; s += 256) {
    const float* kp = k2ws + ((size_t)s * NN + b) * EE;
    float d = 0.f;
    #pragma unroll 8
    for (int e = 0; e < EE; ++e) d += qp[e] * kp[e];
    sc[s] = d;
  }
  __syncthreads();

  // softmax over s
  red[t] = fmaxf(sc[t], sc[t + 256]);
  __syncthreads();
  for (int o = 128; o >= 1; o >>= 1) {
    if (t < o) red[t] = fmaxf(red[t], red[t + o]);
    __syncthreads();
  }
  float mx = red[0];
  __syncthreads();
  float e0 = __expf(sc[t] - mx), e1 = __expf(sc[t + 256] - mx);
  sc[t] = e0; sc[t + 256] = e1;
  red[t] = e0 + e1;
  __syncthreads();
  for (int o = 128; o >= 1; o >>= 1) {
    if (t < o) red[t] += red[t + o];
    __syncthreads();
  }
  float inv = 1.0f / red[0];
  __syncthreads();

  // ctx[e] = sum_s attn[s] * v2[s][b][e]
  int e = t & 63, g = t >> 6;
  float acc = 0.f;
  const float* vb = v2ws + (size_t)b * EE + e;
  for (int s = g * 128; s < g * 128 + 128; ++s)
    acc += sc[s] * vb[(size_t)s * NN * EE];
  part[g][e] = acc * inv;
  __syncthreads();
  if (t < 64) ctxv[t] = part[0][t] + part[1][t] + part[2][t] + part[3][t];
  __syncthreads();

  // attn_out[0] = ctx @ W_out^T + b_out
  if (t < 64) {
    float a = b_out[t];
    const float* wp = W_out + (size_t)t * EE;
    #pragma unroll 8
    for (int x = 0; x < EE; ++x) a += wp[x] * ctxv[x];
    red[t] = a;
  }
  __syncthreads();

  // out = attn_vectors @ W_o^T + b_o
  if (t < OUTD) {
    float a = b_o[t];
    const float* wp = W_o + (size_t)t * EE;
    #pragma unroll 8
    for (int x = 0; x < EE; ++x) a += wp[x] * red[x];
    out[(size_t)b * OUTD + t] = a;
  }
}

// ---------------------------------------------------------------------------
// Launch
// ---------------------------------------------------------------------------
extern "C" void kernel_launch(void* const* d_in, const int* in_sizes, int n_in,
                              void* d_out, int out_size, void* d_ws, size_t ws_size,
                              hipStream_t stream) {
  const float* hidden_state = (const float*)d_in[0];
  const float* obs1  = (const float*)d_in[1];
  const float* obs2  = (const float*)d_in[2];
  const float* h0    = (const float*)d_in[3];
  const float* c0    = (const float*)d_in[4];
  const float* W_s   = (const float*)d_in[5];
  const float* b_s   = (const float*)d_in[6];
  const float* W_v   = (const float*)d_in[7];
  const float* b_v   = (const float*)d_in[8];
  const float* W_hs  = (const float*)d_in[9];
  const float* b_hs  = (const float*)d_in[10];
  const float* W_ih  = (const float*)d_in[11];
  const float* b_ih  = (const float*)d_in[12];
  const float* W_hh  = (const float*)d_in[13];
  const float* b_hh  = (const float*)d_in[14];
  const float* Wq    = (const float*)d_in[15];
  const float* Wk    = (const float*)d_in[16];
  const float* Wv    = (const float*)d_in[17];
  const float* W_in  = (const float*)d_in[18];
  const float* b_in  = (const float*)d_in[19];
  const float* W_out = (const float*)d_in[20];
  const float* b_out = (const float*)d_in[21];
  const float* W_o   = (const float*)d_in[22];
  const float* b_o   = (const float*)d_in[23];
  float* out = (float*)d_out;

  char* ws = (char*)d_ws;
  _Float16* BgT  = (_Float16*)(ws + OFF_BGT);
  _Float16* BkvT = (_Float16*)(ws + OFF_BKVT);
  float*    hs   = (float*)(ws + OFF_HS);
  float*    q0   = (float*)(ws + OFF_Q0);
  float*    k2ws = (float*)(ws + OFF_K2);
  float*    v2ws = (float*)(ws + OFF_V2);

  hipLaunchKernelGGL(dsa_prep, dim3(1), dim3(256), 0, stream,
                     hidden_state, W_hs, b_hs, W_ih, W_hh, Wq, Wk, Wv,
                     W_in, b_in, BgT, BkvT, hs, q0, k2ws, v2ws);

  hipLaunchKernelGGL(dsa_main, dim3(MAIN_BLOCKS), dim3(256), 0, stream,
                     obs1, obs2, h0, c0, W_s, b_s, W_v, b_v,
                     b_ih, b_hh, b_in, BgT, BkvT, k2ws, v2ws);

  hipLaunchKernelGGL(dsa_attn, dim3(NN), dim3(256), 0, stream,
                     q0, k2ws, v2ws, W_out, b_out, W_o, b_o, out);
}
